// SpikingNeuralNetwork_42691974922958
// MI455X (gfx1250) — matrix operand order repaired and checked by reading, not compiled, not run on hardware
//
#include <hip/hip_runtime.h>

// ---------------------------------------------------------------------------
// Spiking NN: 100 time steps of  LIF -> GEMM1(bf16 WMMA) -> LIF ->
// GEMM2(bf16 WMMA) -> LIF+acc, with STDP update of W1 every 10 steps.
// Dims fixed by reference setup: B=128, In=1024, H=4096, Out=1024, T=100.
// GEMM: per-block LDS staging of the shared B tile via gfx1250 async
// global->LDS copies (ASYNCcnt), double buffered under the WMMAs.
// ---------------------------------------------------------------------------

#define B_   128
#define IN_  1024
#define H_   4096
#define OUT_ 1024
#define T_   100

typedef __attribute__((ext_vector_type(16))) __bf16 v16bf;
typedef __attribute__((ext_vector_type(8)))  __bf16 v8bf;
typedef __attribute__((ext_vector_type(8)))  float  v8f;

union Frag16 { v16bf v; v8bf h[2]; };

// ---------------- CDNA5 async global->LDS copy helpers ---------------------
__device__ __forceinline__ void async_load_b128(unsigned lds_off, const void* gaddr) {
    // GLOBAL_LOAD_ASYNC_TO_LDS_B128 (GV mode): VDST = LDS byte offset,
    // VADDR = 64-bit global address. Tracked with ASYNCcnt.
    asm volatile("global_load_async_to_lds_b128 %0, %1, off"
                 :
                 : "v"(lds_off), "v"((unsigned long long)(uintptr_t)gaddr)
                 : "memory");
}
__device__ __forceinline__ void wait_asynccnt0() {
    asm volatile("s_wait_asynccnt 0" ::: "memory");
}

// ---------------- LIF ------------------------------------------------------
__device__ __forceinline__ void lif_step(float& v, float& r, float I, float& spk) {
    bool active = I > 0.0f;
    bool refrac = r > 0.0f;
    float v_int = v * 0.95f + I;
    bool fired  = active && !refrac && (v_int >= 1.0f);
    float v_new = !active ? v : ((refrac || fired) ? 0.0f : v_int);
    float r_new = !active ? r : (refrac ? (r - 1.0f) : (fired ? 2.0f : r));
    v = v_new; r = r_new; spk = fired ? 1.0f : 0.0f;
}

// ---------------- init kernels --------------------------------------------
__global__ void k_w1_init(const float* __restrict__ W1, float* __restrict__ W1f,
                          __bf16* __restrict__ W1b, int n) {
    int i = blockIdx.x * 256 + threadIdx.x;
    if (i >= n) return;
    float w = W1[i];
    W1f[i] = w;
    W1b[i] = (__bf16)w;
}

__global__ void k_w2_init(const float* __restrict__ W2, __bf16* __restrict__ W2b, int n) {
    int i = blockIdx.x * 256 + threadIdx.x;
    if (i >= n) return;
    W2b[i] = (__bf16)W2[i];
}

__global__ void k_zero_f(float* __restrict__ p, int n) {
    int i = blockIdx.x * 256 + threadIdx.x;
    if (i < n) p[i] = 0.0f;
}

__global__ void k_zero_i(int* __restrict__ p, int n) {
    int i = blockIdx.x * 256 + threadIdx.x;
    if (i < n) p[i] = 0;
}

// ---------------- per-step elementwise kernels -----------------------------
__global__ void k_input_lif(const float* __restrict__ x, const float* __restrict__ noise,
                            float* __restrict__ v, float* __restrict__ r,
                            __bf16* __restrict__ spk, int t, int n) {
    int i = blockIdx.x * 256 + threadIdx.x;
    if (i >= n) return;
    float prob = fminf(fmaxf(x[i] * 100.0f, 0.0f), 100.0f) * 0.001f;  // rate * DT/1000
    float I = (noise[(size_t)t * n + i] < prob) ? 1.0f : 0.0f;
    float vv = v[i], rr = r[i], s;
    lif_step(vv, rr, I, s);
    v[i] = vv; r[i] = rr; spk[i] = (__bf16)s;
}

__global__ void k_hidden_lif(const float* __restrict__ I, float* __restrict__ v,
                             float* __restrict__ r, __bf16* __restrict__ spk, int n) {
    int i = blockIdx.x * 256 + threadIdx.x;
    if (i >= n) return;
    float vv = v[i], rr = r[i], s;
    lif_step(vv, rr, I[i], s);
    v[i] = vv; r[i] = rr; spk[i] = (__bf16)s;
}

__global__ void k_output_lif(const float* __restrict__ I, float* __restrict__ v,
                             float* __restrict__ r, float* __restrict__ acc, int n) {
    int i = blockIdx.x * 256 + threadIdx.x;
    if (i >= n) return;
    float vv = v[i], rr = r[i], s;
    lif_step(vv, rr, I[i], s);
    v[i] = vv; r[i] = rr; acc[i] += s;
}

__global__ void k_final(const float* __restrict__ acc, float* __restrict__ out, int n) {
    int i = blockIdx.x * 256 + threadIdx.x;
    if (i < n) out[i] = acc[i] * (1.0f / (float)T_);
}

// ---------------- bf16 WMMA GEMM:  C[MxN] = A[MxK] * W[NxK]^T --------------
// One block (8 waves) -> full M=128 x 64(N) tile; blockIdx.x selects N tile.
// Wave w computes rows [16w,16w+16). The 64x32 B tile is shared by all 8
// waves: staged into LDS with async global->LDS b128 copies, double buffered.
#define BROW_PAD 40   // 32 bf16 + 8 pad (80B row stride) to spread LDS banks

__global__ __launch_bounds__(256)
void k_gemm_bf16(const __bf16* __restrict__ A, const __bf16* __restrict__ W,
                 float* __restrict__ C, int M, int N, int K) {
    __shared__ __align__(16) __bf16 Bs[2][64][BROW_PAD];

    const int tid  = threadIdx.x;
    const int wv   = tid >> 5;                 // wave 0..7 -> mTile
    const int lane = tid & 31;
    const int m0 = wv << 4;
    const int n0 = blockIdx.x << 6;            // 64-wide N tile
    const int half = lane >> 4;                // 0: lanes 0-15, 1: lanes 16-31
    const int lrow = lane & 15;

    // staging assignment: thread -> (row 0..63, 16B k-slice 0..3)
    const int sRow = tid >> 2;
    const int sKp  = (tid & 3) << 3;           // bf16 element offset (0,8,16,24)
    const __bf16* wSrc = W + (size_t)(n0 + sRow) * K + sKp;

    v8f acc0 = {}, acc1 = {}, acc2 = {}, acc3 = {};
    const __bf16* aRow = A + (size_t)(m0 + lrow) * K + half * 16;

    const int KC = K >> 5;                     // 32-wide K chunks

    // preload chunk 0 into buffer 0
    async_load_b128((unsigned)(uintptr_t)&Bs[0][sRow][sKp], wSrc);

    for (int kc = 0; kc < KC; ++kc) {
        const int cur = kc & 1;
        const int k0 = kc << 5;

        wait_asynccnt0();                      // my async store to LDS landed
        __syncthreads();                       // everyone's landed; prev reads done

        if (kc + 1 < KC) {                     // stage next chunk into other buffer
            async_load_b128((unsigned)(uintptr_t)&Bs[cur ^ 1][sRow][sKp],
                            wSrc + (size_t)(k0 + 32));
        }

        Frag16 a;
        const v8bf* pa = (const v8bf*)(aRow + k0);
        a.h[0] = pa[0];                        // K = k0+16h   .. +7
        a.h[1] = pa[1];                        // K = k0+16h+8 .. +15
        __builtin_prefetch(aRow + k0 + 64, 0, 3);

        Frag16 b;
        const v8bf* pb;
        pb = (const v8bf*)&Bs[cur][lrow][16 * half];
        b.h[0] = pb[0]; b.h[1] = pb[1];
        acc0 = __builtin_amdgcn_wmma_f32_16x16x32_bf16(false, a.v, false, b.v,
                                                       (short)0, acc0, false, false);
        pb = (const v8bf*)&Bs[cur][16 + lrow][16 * half];
        b.h[0] = pb[0]; b.h[1] = pb[1];
        acc1 = __builtin_amdgcn_wmma_f32_16x16x32_bf16(false, a.v, false, b.v,
                                                       (short)0, acc1, false, false);
        pb = (const v8bf*)&Bs[cur][32 + lrow][16 * half];
        b.h[0] = pb[0]; b.h[1] = pb[1];
        acc2 = __builtin_amdgcn_wmma_f32_16x16x32_bf16(false, a.v, false, b.v,
                                                       (short)0, acc2, false, false);
        pb = (const v8bf*)&Bs[cur][48 + lrow][16 * half];
        b.h[0] = pb[0]; b.h[1] = pb[1];
        acc3 = __builtin_amdgcn_wmma_f32_16x16x32_bf16(false, a.v, false, b.v,
                                                       (short)0, acc3, false, false);
    }

    // D layout: VGPR r -> row m0 + half*8 + r, column n0 + lrow (+16j)
    float* crow = C + (size_t)(m0 + half * 8) * N + n0 + lrow;
#pragma unroll
    for (int rr = 0; rr < 8; ++rr) {
        crow[(size_t)rr * N +  0] = acc0[rr];
        crow[(size_t)rr * N + 16] = acc1[rr];
        crow[(size_t)rr * N + 32] = acc2[rr];
        crow[(size_t)rr * N + 48] = acc3[rr];
    }
}

// ---------------- STDP helper kernels --------------------------------------
__global__ void k_colmean(const __bf16* __restrict__ S, float* __restrict__ mean,
                          int rows, int cols, int* __restrict__ flag) {
    int c = blockIdx.x * 256 + threadIdx.x;
    if (c >= cols) return;
    float s = 0.0f;
    for (int r = 0; r < rows; ++r) s += (float)S[(size_t)r * cols + c];
    mean[c] = s / (float)rows;
    if (s > 0.0f) atomicOr(flag, 1);
}

__global__ void k_trace(float* __restrict__ tp, const float* __restrict__ pre, int nIn,
                        float* __restrict__ tn, const float* __restrict__ post, int nH) {
    int i = blockIdx.x * 256 + threadIdx.x;
    if (i < nIn) tp[i] = 0.9f * tp[i] + pre[i];
    if (i < nH)  tn[i] = 0.9f * tn[i] + post[i];
}

__global__ void k_w1_update(float* __restrict__ W1f, __bf16* __restrict__ W1b,
                            const float* __restrict__ tp, const float* __restrict__ tn,
                            const float* __restrict__ pre, const float* __restrict__ post,
                            const int* __restrict__ flags, int n) {
    int idx = blockIdx.x * 256 + threadIdx.x;
    if (idx >= n) return;
    if (flags[0] == 0 || flags[1] == 0) return;     // both = any in-spike & any hid-spike
    int i = idx & (IN_ - 1);
    int h = idx >> 10;                               // IN_ == 1024
    float dW = 0.001f * (tp[i] * post[h] - pre[i] * tn[h]);
    float w = W1f[idx] + dW;
    w = fminf(fmaxf(w, -1.0f), 1.0f);
    W1f[idx] = w;
    W1b[idx] = (__bf16)w;
}

// ---------------------------------------------------------------------------
extern "C" void kernel_launch(void* const* d_in, const int* in_sizes, int n_in,
                              void* d_out, int out_size, void* d_ws, size_t ws_size,
                              hipStream_t stream) {
    const float* x     = (const float*)d_in[0];   // (B, In)
    const float* W1    = (const float*)d_in[1];   // (H, In)
    const float* W2    = (const float*)d_in[2];   // (Out, H)
    const float* noise = (const float*)d_in[3];   // (T, B, In)
    (void)in_sizes; (void)n_in; (void)out_size; (void)ws_size;

    // ---- carve workspace (256B aligned slabs) ----
    char* p = (char*)d_ws;
    auto carve = [&](size_t bytes) -> void* {
        void* r = (void*)p;
        p += (bytes + 255) & ~(size_t)255;
        return r;
    };
    const int   W1N = H_ * IN_;           // 4194304
    const int   W2N = OUT_ * H_;          // 4194304
    float*  W1f  = (float*)carve((size_t)W1N * 4);
    __bf16* W1b  = (__bf16*)carve((size_t)W1N * 2);
    __bf16* W2b  = (__bf16*)carve((size_t)W2N * 2);
    // zero-initialized state block (contiguous)
    const int nBI = B_ * IN_, nBH = B_ * H_, nBO = B_ * OUT_;
    const int zeroN = 2 * nBI + 2 * nBH + 2 * nBO + nBO + IN_ + H_;
    float* zbase = (float*)carve((size_t)zeroN * 4);
    float* v_i = zbase;
    float* r_i = v_i + nBI;
    float* v_h = r_i + nBI;
    float* r_h = v_h + nBH;
    float* v_o = r_h + nBH;
    float* r_o = v_o + nBO;
    float* acc = r_o + nBO;
    float* tp  = acc + nBO;
    float* tn  = tp + IN_;
    // non-persistent scratch
    __bf16* in_spk  = (__bf16*)carve((size_t)nBI * 2);
    __bf16* hid_spk = (__bf16*)carve((size_t)nBH * 2);
    float*  hid_I   = (float*)carve((size_t)nBH * 4);
    float*  out_I   = (float*)carve((size_t)nBO * 4);
    float*  pre_m   = (float*)carve((size_t)IN_ * 4);
    float*  post_m  = (float*)carve((size_t)H_ * 4);
    int*    flags   = (int*)carve(2 * sizeof(int));

    // ---- init (fresh state every call: deterministic) ----
    k_w1_init<<<(W1N + 255) / 256, 256, 0, stream>>>(W1, W1f, W1b, W1N);
    k_w2_init<<<(W2N + 255) / 256, 256, 0, stream>>>(W2, W2b, W2N);
    k_zero_f<<<(zeroN + 255) / 256, 256, 0, stream>>>(zbase, zeroN);

    const int gBI = nBI / 256, gBH = nBH / 256, gBO = nBO / 256;
    const int gemm1Blocks = H_ / 64;     // one block per 64-wide N tile
    const int gemm2Blocks = OUT_ / 64;

    for (int t = 0; t < T_; ++t) {
        k_input_lif<<<gBI, 256, 0, stream>>>(x, noise, v_i, r_i, in_spk, t, nBI);
        k_gemm_bf16<<<gemm1Blocks, 256, 0, stream>>>(in_spk, W1b, hid_I, B_, H_, IN_);
        k_hidden_lif<<<gBH, 256, 0, stream>>>(hid_I, v_h, r_h, hid_spk, nBH);
        k_gemm_bf16<<<gemm2Blocks, 256, 0, stream>>>(hid_spk, W2b, out_I, B_, OUT_, H_);
        k_output_lif<<<gBO, 256, 0, stream>>>(out_I, v_o, r_o, acc, nBO);

        if ((t % 10) == 0) {
            k_zero_i<<<1, 256, 0, stream>>>(flags, 2);
            k_colmean<<<(IN_ + 255) / 256, 256, 0, stream>>>(in_spk, pre_m, B_, IN_, flags + 0);
            k_colmean<<<(H_ + 255) / 256, 256, 0, stream>>>(hid_spk, post_m, B_, H_, flags + 1);
            k_trace<<<(H_ + 255) / 256, 256, 0, stream>>>(tp, pre_m, IN_, tn, post_m, H_);
            k_w1_update<<<(W1N + 255) / 256, 256, 0, stream>>>(W1f, W1b, tp, tn,
                                                               pre_m, post_m, flags, W1N);
        }
    }

    k_final<<<gBO, 256, 0, stream>>>(acc, (float*)d_out, nBO);
}
